// SelectiveSampler_38800734552241
// MI455X (gfx1250) — matrix operand order repaired and checked by reading, not compiled, not run on hardware
//
#include <hip/hip_runtime.h>

// ---------------------------------------------------------------------------
// SelectiveSampler for MI455X (gfx1250, wave32)
//
// Pipeline (D=160, N=D^3=4,096,000, all L2-resident on MI455X's 192MB L2):
//  1) k_init       : round(data), seg fields (bf16 0/1) for v=0,1,2; CCL seed
//  2) 16x k_ccl    : Jacobi 6-neighbor min-propagation (only v=1 is needed:
//                    means for v=0,2 are discarded by the reference)
//  3) k_count      : component sizes via u32 atomic scatter (deterministic)
//  4) box filters  : X-axis pass = banded 0/1 matmul done EXACTLY with
//                    v_wmma_f32_16x16x32_bf16 (0/1 inputs exact in bf16,
//                    f32 accumulate); Y/Z (and the large-valued cs pass)
//                    use exact running-window sweeps.
//  5) reciprocal-histogram weighting collapses algebraically to
//                    proba[i] = 1 / (nb * hist[idx_i])   (idx>=1), else 0
//     where nb = #nonzero bins in [1,4096). Exactly equivalent to reference.
// ---------------------------------------------------------------------------

#define DSIDE  160
#define PLANE  25600            // 160*160
#define NVOX   4096000          // 160^3
#define BIGLAB 4096002          // n_vox + 2
#define NUMBINS 4096

typedef __attribute__((ext_vector_type(16))) __bf16 v16bf;
typedef __attribute__((ext_vector_type(8)))  __bf16 v8bf;
typedef __attribute__((ext_vector_type(8)))  float  v8f;

// ---------------------------------------------------------------- init ------
__global__ __launch_bounds__(256) void k_init(const float* __restrict__ data,
                                              __bf16* __restrict__ s0,
                                              __bf16* __restrict__ s1,
                                              __bf16* __restrict__ s2,
                                              int* __restrict__ lab) {
    int i = blockIdx.x * 256 + threadIdx.x;
    if (i >= NVOX) return;
    float r = rintf(data[i]);                       // jnp.round (RNE)
    s0[i] = (__bf16)(r == 0.0f ? 1.0f : 0.0f);
    s1[i] = (__bf16)(r == 1.0f ? 1.0f : 0.0f);
    s2[i] = (__bf16)(r == 2.0f ? 1.0f : 0.0f);
    lab[i] = (r == 1.0f) ? (i + 1) : BIGLAB;        // seg ? idx+1 : big
}

// ---------------------------------------------------------------- CCL -------
__global__ __launch_bounds__(256) void k_ccl(const int* __restrict__ lab,
                                             int* __restrict__ out) {
    int i = blockIdx.x * 256 + threadIdx.x;
    if (i >= NVOX) return;
    int v = lab[i];
    if (v == BIGLAB) { out[i] = BIGLAB; return; }   // non-seg stays big
    int x = i % DSIDE;
    int t = i / DSIDE;
    int y = t % DSIDE;
    int z = t / DSIDE;
    int m = v;
    if (x > 0)         m = min(m, lab[i - 1]);
    if (x < DSIDE - 1) m = min(m, lab[i + 1]);
    if (y > 0)         m = min(m, lab[i - DSIDE]);
    if (y < DSIDE - 1) m = min(m, lab[i + DSIDE]);
    if (z > 0)         m = min(m, lab[i - PLANE]);
    if (z < DSIDE - 1) m = min(m, lab[i + PLANE]);
    out[i] = m;
}

// ------------------------------------------------------------- counting -----
__global__ __launch_bounds__(256) void k_zero_u32(unsigned* __restrict__ p, int n) {
    int i = blockIdx.x * 256 + threadIdx.x;
    if (i < n) p[i] = 0u;
}

__global__ __launch_bounds__(256) void k_count(const int* __restrict__ lab,
                                               unsigned* __restrict__ cnt) {
    int i = blockIdx.x * 256 + threadIdx.x;
    if (i >= NVOX) return;
    int l = lab[i];
    int bin = (l <= NVOX) ? l : 0;                  // background -> 0
    atomicAdd(&cnt[bin], 1u);
}

// ------------------------------------------------ X-axis box sum via WMMA ---
// out(25600 x 160) = M(25600 x 160) x Band(160 x 160),
// Band[k,n] = 1 iff n-16 <= k <= n+15 (window [i-16, i+15], zeros outside).
// One wave per 16-row tile; band sparsity => <=2 nonzero K-chunks / col tile.
__global__ __launch_bounds__(256) void k_box_x_wmma(const __bf16* __restrict__ seg,
                                                    float* __restrict__ out) {
    const int wave    = blockIdx.x * 8 + (threadIdx.x >> 5);   // 1600 row tiles
    const int lane    = threadIdx.x & 31;
    const int l15     = lane & 15;
    const bool hi     = lane >= 16;
    const int rowBase = wave * 16;
    const int m       = rowBase + l15;                          // A row of lane
    const __bf16* arow = seg + (size_t)m * DSIDE;

    // A fragments, documented 16-bit A 16x32 layout:
    //  lo lanes: K = kBase+{0..7} in V0..3, kBase+{16..23} in V4..7
    //  hi lanes: same shifted by +8
    v16bf afrag[5];
#pragma unroll
    for (int kc = 0; kc < 5; ++kc) {
        int kb = kc * 32 + (hi ? 8 : 0);
        v8bf lo = *(const v8bf*)(arow + kb);        // 16B aligned (stride 320B)
        v8bf h2 = *(const v8bf*)(arow + kb + 16);
        v16bf a;
#pragma unroll
        for (int e = 0; e < 8; ++e) { a[e] = lo[e]; a[8 + e] = h2[e]; }
        afrag[kc] = a;
    }

#pragma unroll
    for (int nT = 0; nT < 10; ++nT) {
        const int colBase = nT * 16;
        const int n = colBase + l15;                // D/B column of this lane
        v8f acc = {0.f, 0.f, 0.f, 0.f, 0.f, 0.f, 0.f, 0.f};
#pragma unroll
        for (int kc = 0; kc < 5; ++kc) {
            // compile-time band-intersection cull (nT,kc constants after unroll)
            if (32 * kc + 31 < colBase - 16) continue;
            if (32 * kc > colBase + 15 + 15) continue;
            // B fragment: lane holds col n; element e <-> row 32kc+(hi?8:0)+(e<8?e:e+8)
            v16bf b;
#pragma unroll
            for (int e = 0; e < 16; ++e) {
                int k = 32 * kc + (hi ? 8 : 0) + (e < 8 ? e : e + 8);
                int d = k - n + 16;
                b[e] = (__bf16)((d >= 0 && d < 32) ? 1.0f : 0.0f);
            }
            acc = __builtin_amdgcn_wmma_f32_16x16x32_bf16(
                false, afrag[kc], false, b, (short)0, acc, false, false);
        }
        // D layout: VGPR r, lo lanes -> row rowBase+r, hi lanes -> rowBase+8+r
        float* o = out + (size_t)(rowBase + (hi ? 8 : 0)) * DSIDE + n;
#pragma unroll
        for (int r = 0; r < 8; ++r) o[(size_t)r * DSIDE] = acc[r];
    }
}

// ----------------------------------------- running-window 1D box (Y/Z/X) ----
__global__ __launch_bounds__(256) void k_box_line(const float* __restrict__ in,
                                                  float* __restrict__ out,
                                                  int axis) {
    int l = blockIdx.x * 256 + threadIdx.x;
    if (l >= PLANE) return;
    size_t base, stride;
    if (axis == 0)      { base = (size_t)l * DSIDE; stride = 1; }
    else if (axis == 1) { int z = l / DSIDE, x = l % DSIDE;
                          base = (size_t)z * PLANE + x; stride = DSIDE; }
    else                { base = (size_t)l; stride = PLANE; }
    const float* p = in + base;
    float* q = out + base;
    float s = 0.f;
#pragma unroll
    for (int j = 0; j < 16; ++j) s += p[(size_t)j * stride];
    q[0] = s;                                       // window [-16,15] clipped
    for (int i = 1; i < DSIDE; ++i) {
        int add = i + 15, sub = i - 16;
        if (add < DSIDE) s += p[(size_t)add * stride];
        if (sub >= 0)    s -= p[(size_t)sub * stride];
        q[(size_t)i * stride] = s;
    }
}

// --------------------------------------------------------- valid accum ------
__global__ __launch_bounds__(256) void k_valid(const float* __restrict__ cw,
                                               float* __restrict__ acc, int add) {
    int i = blockIdx.x * 256 + threadIdx.x;
    if (i >= NVOX) return;
    float v = (cw[i] > 100.0f) ? 1.0f : 0.0f;       // cw > max(NUM_MIN,1)
    acc[i] = add ? (acc[i] + v) : v;
}

// --------------------------------------- candidates + lsize*valid field -----
__global__ __launch_bounds__(256) void k_cand(const float* __restrict__ cw1,
                                              const float* __restrict__ acc,
                                              const int* __restrict__ lab,
                                              const unsigned* __restrict__ cnt,
                                              float* __restrict__ outCand,
                                              float* __restrict__ lsval) {
    int i = blockIdx.x * 256 + threadIdx.x;
    if (i >= NVOX) return;
    int x = i % DSIDE;
    int t = i / DSIDE;
    int y = t % DSIDE;
    int z = t / DSIDE;
    float v1 = (cw1[i] > 100.0f) ? 1.0f : 0.0f;
    bool inner = (x >= 16 && x < 144 && y >= 16 && y < 144 && z >= 16 && z < 144);
    float fin = (inner ? v1 : 0.0f) + acc[i];       // pad-ones * valid1 + others
    outCand[i] = (fin >= 2.0f) ? 1.0f : 0.0f;       // >= num_labels_add + 1
    int l = lab[i];
    lsval[i] = (l <= NVOX && v1 > 0.0f) ? (float)cnt[l] : 0.0f;
}

// ------------------------------------------------------ idx + histogram -----
__global__ __launch_bounds__(256) void k_idxhist(const float* __restrict__ cs,
                                                 const float* __restrict__ cw1,
                                                 const float* __restrict__ cand,
                                                 int* __restrict__ idxArr,
                                                 unsigned* __restrict__ hist) {
    int i = blockIdx.x * 256 + threadIdx.x;
    if (i >= NVOX) return;
    float cw = cw1[i];
    float mean = (cw > 0.0f) ? (cs[i] / fmaxf(cw, 1.0f)) : 0.0f;
    float p = cand[i] * mean;
    float r = rintf(p);                             // jnp.round (RNE)
    r = fminf(fmaxf(r, 0.0f), (float)(NUMBINS - 1));
    int idx = (int)r;
    idxArr[i] = idx;
    atomicAdd(&hist[idx], 1u);
}

// --------------------------------------------------- nonzero-bin count ------
__global__ __launch_bounds__(256) void k_nb(const unsigned* __restrict__ hist,
                                            unsigned* __restrict__ nbOut) {
    __shared__ unsigned sh[256];
    unsigned c = 0;
    for (int b = threadIdx.x; b < NUMBINS; b += 256)
        if (b >= 1 && hist[b] > 0u) ++c;
    sh[threadIdx.x] = c;
    __syncthreads();
    for (int s = 128; s > 0; s >>= 1) {
        if ((int)threadIdx.x < s) sh[threadIdx.x] += sh[threadIdx.x + s];
        __syncthreads();
    }
    if (threadIdx.x == 0) nbOut[0] = sh[0];
}

// ---------------------------------------------------------- final proba -----
// proba = ph_full[idx] / sum(pw) == 1 / (nb * hist[idx]) for idx>=1, else 0.
__global__ __launch_bounds__(256) void k_proba(const int* __restrict__ idxArr,
                                               const unsigned* __restrict__ hist,
                                               const unsigned* __restrict__ nbPtr,
                                               float* __restrict__ outP) {
    int i = blockIdx.x * 256 + threadIdx.x;
    if (i >= NVOX) return;
    int idx = idxArr[i];
    float nb = (float)nbPtr[0];
    outP[i] = (idx >= 1) ? (1.0f / (nb * (float)hist[idx])) : 0.0f;
}

// ---------------------------------------------------------------------------
extern "C" void kernel_launch(void* const* d_in, const int* in_sizes, int n_in,
                              void* d_out, int out_size, void* d_ws, size_t ws_size,
                              hipStream_t stream) {
    const float* data = (const float*)d_in[0];
    float* outCand = (float*)d_out;                 // first N floats
    float* outProb = (float*)d_out + NVOX;          // second N floats
    float* valAcc  = outProb;                       // reused as scratch (written
                                                    // before read each call)

    // carve workspace (256B aligned), total ~123 MB
    char* w = (char*)d_ws;
    auto carve = [&](size_t bytes) -> void* {
        void* p = (void*)w;
        w += (bytes + 255) & ~(size_t)255;
        return p;
    };
    __bf16* s0     = (__bf16*)  carve((size_t)NVOX * 2);
    __bf16* s1     = (__bf16*)  carve((size_t)NVOX * 2);
    __bf16* s2     = (__bf16*)  carve((size_t)NVOX * 2);
    int*    lab    = (int*)     carve((size_t)NVOX * 4);
    int*    labT   = (int*)     carve((size_t)NVOX * 4);
    unsigned* cnt  = (unsigned*)carve(((size_t)NVOX + 1) * 4);
    float*  bufA   = (float*)   carve((size_t)NVOX * 4);
    float*  bufB   = (float*)   carve((size_t)NVOX * 4);
    float*  cw1    = (float*)   carve((size_t)NVOX * 4);
    unsigned* hist = (unsigned*)carve((size_t)NUMBINS * 4);
    unsigned* nb   = (unsigned*)carve(256);

    const int NBLK   = (NVOX + 255) / 256;          // 16000
    const int NBLK1  = (NVOX + 1 + 255) / 256;
    const int LBLK   = (PLANE + 255) / 256;         // 100
    const int WBLK   = 200;                         // 1600 waves / 8 per block

    // 1) segmentation + CCL seed
    k_init<<<NBLK, 256, 0, stream>>>(data, s0, s1, s2, lab);

    // 2) 16 min-propagation sweeps (ping-pong, even count -> result in lab)
    for (int s = 0; s < 16; ++s) {
        if ((s & 1) == 0) k_ccl<<<NBLK, 256, 0, stream>>>(lab, labT);
        else              k_ccl<<<NBLK, 256, 0, stream>>>(labT, lab);
    }

    // 3) component sizes
    k_zero_u32<<<NBLK1, 256, 0, stream>>>(cnt, NVOX + 1);
    k_count<<<NBLK, 256, 0, stream>>>(lab, cnt);

    // 4) box filters. X pass via WMMA band-matmul, Y/Z via running window.
    // v = 0
    k_box_x_wmma<<<WBLK, 256, 0, stream>>>(s0, bufA);
    k_box_line<<<LBLK, 256, 0, stream>>>(bufA, bufB, 1);
    k_box_line<<<LBLK, 256, 0, stream>>>(bufB, bufA, 2);
    k_valid<<<NBLK, 256, 0, stream>>>(bufA, valAcc, 0);
    // v = 2
    k_box_x_wmma<<<WBLK, 256, 0, stream>>>(s2, bufA);
    k_box_line<<<LBLK, 256, 0, stream>>>(bufA, bufB, 1);
    k_box_line<<<LBLK, 256, 0, stream>>>(bufB, bufA, 2);
    k_valid<<<NBLK, 256, 0, stream>>>(bufA, valAcc, 1);
    // v = 1 (keep counts window)
    k_box_x_wmma<<<WBLK, 256, 0, stream>>>(s1, bufA);
    k_box_line<<<LBLK, 256, 0, stream>>>(bufA, bufB, 1);
    k_box_line<<<LBLK, 256, 0, stream>>>(bufB, cw1, 2);

    // 5) candidates + lsize*valid field (into bufA)
    k_cand<<<NBLK, 256, 0, stream>>>(cw1, valAcc, lab, cnt, outCand, bufA);

    // 6) counts_size = box_sum(lsize*valid): exact running sums (large values)
    k_box_line<<<LBLK, 256, 0, stream>>>(bufA, bufB, 0);
    k_box_line<<<LBLK, 256, 0, stream>>>(bufB, bufA, 1);
    k_box_line<<<LBLK, 256, 0, stream>>>(bufA, bufB, 2);   // cs in bufB

    // 7) mean -> bin index + histogram (idx stored in labT, now free)
    k_zero_u32<<<(NUMBINS + 255) / 256, 256, 0, stream>>>(hist, NUMBINS);
    k_idxhist<<<NBLK, 256, 0, stream>>>(bufB, cw1, outCand, labT, hist);

    // 8) closed-form reciprocal-histogram weights
    k_nb<<<1, 256, 0, stream>>>(hist, nb);
    k_proba<<<NBLK, 256, 0, stream>>>(labT, hist, nb, outProb);
    (void)in_sizes; (void)n_in; (void)out_size; (void)ws_size;
}
